// SelfAttention_88957362635341
// MI455X (gfx1250) — matrix operand order repaired and checked by reading, not compiled, not run on hardware
//
#include <hip/hip_runtime.h>
#include <hip/hip_bf16.h>

// ---------------------------------------------------------------------------
// Self-attention forward on gfx1250 (MI455X), wave32:
//   - v_wmma_f32_16x16x32_f16 for all four GEMM stages
//   - global_load_async_to_lds_b128 (ASYNCcnt) for K/V and Wo LDS staging
// ---------------------------------------------------------------------------

#define N_B 2
#define L_S 2048
#define H_N 16
#define D_H 64
#define E_D 1024

typedef _Float16 v8h  __attribute__((ext_vector_type(8)));
typedef _Float16 v16h __attribute__((ext_vector_type(16)));
typedef float    v8f  __attribute__((ext_vector_type(8)));

// ---- WMMA wrapper: D = A(16x32 f16) * B(32x16 f16) + C(16x16 f32) ----------
__device__ __forceinline__ v8f wmma16x16x32(v16h a, v16h b, v8f c) {
  return __builtin_amdgcn_wmma_f32_16x16x32_f16(
      /*neg_a=*/false, a, /*neg_b=*/false, b,
      /*c_mod=*/(short)0, c, /*reuse_a=*/false, /*reuse_b=*/false);
}

// ---- async global -> LDS copy (16B), tracked by ASYNCcnt -------------------
__device__ __forceinline__ void async_copy_b128(unsigned lds_off, const void* gaddr) {
  asm volatile("global_load_async_to_lds_b128 %0, %1, off"
               :: "v"(lds_off), "v"(gaddr) : "memory");
}
__device__ __forceinline__ void wait_async0() {
  asm volatile("s_wait_asynccnt 0" ::: "memory");
}
__device__ __forceinline__ unsigned lds_offset(const void* p) {
  return (unsigned)(unsigned long long)p;   // generic LDS addr: low 32b = offset
}

// ---- A-fragment loaders (ISA 7.12.2: lane m=lane&15; e<8 -> K=hi*8+e,
//      e>=8 -> K=16+hi*8+(e-8)) ---------------------------------------------
__device__ __forceinline__ v16h load_afrag_f16(const _Float16* row, int ks, int hi) {
  v8h lo = *(const v8h*)(row + ks + hi * 8);
  v8h hh = *(const v8h*)(row + ks + 16 + hi * 8);
  v16h a;
#pragma unroll
  for (int i = 0; i < 8; ++i) { a[i] = lo[i]; a[8 + i] = hh[i]; }
  return a;
}

__device__ __forceinline__ v16h load_afrag_f32(const float* row, int ks, int hi) {
  const float* p0 = row + ks + hi * 8;
  const float* p1 = row + ks + 16 + hi * 8;
  v16h a;
#pragma unroll
  for (int i = 0; i < 8; ++i) { a[i] = (_Float16)p0[i]; a[8 + i] = (_Float16)p1[i]; }
  return a;
}

// ---- B-fragment loader (lane = column n; e<8 -> K=hi*8+e, e>=8 -> K=16+hi*8+(e-8)) ---------
__device__ __forceinline__ v16h load_bfrag_f16(const _Float16* colp, int hi) {
  v8h lo = *(const v8h*)(colp + hi * 8);
  v8h hh = *(const v8h*)(colp + 16 + hi * 8);
  v16h b;
#pragma unroll
  for (int i = 0; i < 8; ++i) { b[i] = lo[i]; b[8 + i] = hh[i]; }
  return b;
}

// ===========================================================================
// Kernel 0: f32 -> f16 array convert (weights, done once)
// ===========================================================================
__global__ void __launch_bounds__(256)
cvt_f16_kernel(const float* __restrict__ src, _Float16* __restrict__ dst, int count) {
  int i = (blockIdx.x * 256 + threadIdx.x) * 4;
  if (i + 3 < count) {
#pragma unroll
    for (int j = 0; j < 4; ++j) dst[i + j] = (_Float16)src[i + j];
  }
}

// ===========================================================================
// Kernel 1: per-head projection from fp32 source:  dst = f16(x @ W^T)
//   src: [N][L][E] fp32 (head h = cols h*64..h*64+63); W16: [64][64] f16
//   dst: [N][H][L][64] f16.  One wave per 16-row tile; 8 WMMAs.
// ===========================================================================
__global__ void __launch_bounds__(128)
proj_f32_kernel(const float* __restrict__ src, const _Float16* __restrict__ W16,
                _Float16* __restrict__ dst) {
  const int wid  = (blockIdx.x * blockDim.x + threadIdx.x) >> 5;
  const int lane = threadIdx.x & 31;
  const int col  = lane & 15, hi = lane >> 4;
  const int lt = wid & 127, h = (wid >> 7) & 15, n = wid >> 11;
  const int l0 = lt * 16;

  const float* arow = src + ((size_t)n * L_S + l0 + col) * E_D + h * D_H;
  v16h a0 = load_afrag_f32(arow, 0, hi);
  v16h a1 = load_afrag_f32(arow, 32, hi);

  _Float16* drow = dst + ((size_t)(n * H_N + h) * L_S + l0) * D_H;
#pragma unroll
  for (int ct = 0; ct < 4; ++ct) {
    const _Float16* w0 = W16 + (size_t)(ct * 16 + col) * D_H;  // column = W row
    v8f c = {};
    c = wmma16x16x32(a0, load_bfrag_f16(w0, hi), c);
    c = wmma16x16x32(a1, load_bfrag_f16(w0 + 32, hi), c);
#pragma unroll
    for (int r = 0; r < 8; ++r)
      drow[(size_t)(hi * 8 + r) * D_H + ct * 16 + col] = (_Float16)c[r];
  }
}

// ===========================================================================
// Kernel 2: v = q16 @ Wv^T, stored TRANSPOSED: vT [N][H][64][L] f16
// ===========================================================================
__global__ void __launch_bounds__(128)
vproj_kernel(const _Float16* __restrict__ q16, const _Float16* __restrict__ Wv16,
             _Float16* __restrict__ vT) {
  const int wid  = (blockIdx.x * blockDim.x + threadIdx.x) >> 5;
  const int lane = threadIdx.x & 31;
  const int col  = lane & 15, hi = lane >> 4;
  const int lt = wid & 127, h = (wid >> 7) & 15, n = wid >> 11;
  const int l0 = lt * 16;

  const _Float16* arow = q16 + ((size_t)(n * H_N + h) * L_S + l0 + col) * D_H;
  v16h a0 = load_afrag_f16(arow, 0, hi);
  v16h a1 = load_afrag_f16(arow, 32, hi);

#pragma unroll
  for (int ct = 0; ct < 4; ++ct) {
    const _Float16* w0 = Wv16 + (size_t)(ct * 16 + col) * D_H;
    v8f c = {};
    c = wmma16x16x32(a0, load_bfrag_f16(w0, hi), c);
    c = wmma16x16x32(a1, load_bfrag_f16(w0 + 32, hi), c);
    v8h pk;
#pragma unroll
    for (int r = 0; r < 8; ++r) pk[r] = (_Float16)c[r];
    // element (key = l0+hi*8+r, d = ct*16+col): 8 contiguous f16 along L
    *(v8h*)(vT + ((size_t)(n * H_N + h) * D_H + ct * 16 + col) * L_S + l0 + hi * 8) = pk;
  }
}

// ===========================================================================
// Kernel 3: flash attention. Block = 4 waves, SAME (n,h), adjacent q-tiles.
// Each 32-key K block (32x64) and V block (64x32, transposed src) is staged
// into LDS via global_load_async_to_lds_b128 and shared by all 4 waves.
// ===========================================================================
__global__ void __launch_bounds__(128)
attn_kernel(const _Float16* __restrict__ q16, const _Float16* __restrict__ k16,
            const _Float16* __restrict__ vT, const int* __restrict__ mask,
            _Float16* __restrict__ attn16) {
  __shared__ _Float16 Kbuf[32 * 64];      // [key][d]
  __shared__ _Float16 Vbuf[64 * 32];      // [d][key]
  __shared__ _Float16 Pbuf[4][16 * 32];   // per-wave P staging

  const int tid   = threadIdx.x;
  const int wslot = tid >> 5;
  const int lane  = tid & 31;
  const int col   = lane & 15, hi = lane >> 4;

  const int b  = blockIdx.x;              // 1024 blocks: n(2) x h(16) x qg(32)
  const int qg = b & 31;
  const int h  = (b >> 5) & 15;
  const int n  = b >> 9;
  const int l0 = (qg * 4 + wslot) * 16;   // this wave's 16 query rows

  const size_t headBase = (size_t)(n * H_N + h) * L_S;   // rows in q16/k16
  const size_t headD    = (size_t)(n * H_N + h) * D_H;   // rows in vT

  const _Float16* arow = q16 + (headBase + l0 + col) * D_H;
  v16h aq0 = load_afrag_f16(arow, 0, hi);
  v16h aq1 = load_afrag_f16(arow, 32, hi);

  float mi[8], li[8];
  v8f acc[4];
#pragma unroll
  for (int r = 0; r < 8; ++r) { mi[r] = -1e30f; li[r] = 0.0f; }
#pragma unroll
  for (int ct = 0; ct < 4; ++ct) acc[ct] = {};

  const float scale = 0.03125f;            // 1/sqrt(1024)
  const int* mrow = mask + n * L_S;
  _Float16* pb = &Pbuf[wslot][0];

  // per-thread async-copy addressing (16 halves = 32B per thread per buffer)
  const unsigned kl_off = lds_offset(Kbuf) + tid * 32;                 // bytes
  const int vd = tid >> 1, vp = tid & 1;
  const unsigned vl_off = lds_offset(Vbuf) + (vd * 32 + vp * 16) * 2;  // bytes

  for (int key0 = 0; key0 < L_S; key0 += 32) {
    __syncthreads();   // previous block's LDS consumers done
    // ---- async stage: K block (contiguous 4KB), V block (64 x 64B) -------
    {
      const _Float16* gk = k16 + (headBase + key0) * D_H + tid * 16;
      async_copy_b128(kl_off,      gk);
      async_copy_b128(kl_off + 16, gk + 8);
      const _Float16* gv = vT + (headD + vd) * L_S + key0 + vp * 16;
      async_copy_b128(vl_off,      gv);
      async_copy_b128(vl_off + 16, gv + 8);
      if (key0 + 32 < L_S) {                                 // prefetch next
        __builtin_prefetch(k16 + (headBase + key0 + 32) * D_H + tid * 16, 0, 1);
        __builtin_prefetch(vT + (headD + vd) * L_S + key0 + 32 + vp * 16, 0, 1);
      }
    }
    wait_async0();
    __syncthreads();

    // ---- S = Q @ K^T for 2 x 16-key tiles (from LDS) ---------------------
    v8f s0 = {}, s1 = {};
    {
      const _Float16* kc0 = Kbuf + (size_t)col * D_H;
      s0 = wmma16x16x32(aq0, load_bfrag_f16(kc0, hi), s0);
      s0 = wmma16x16x32(aq1, load_bfrag_f16(kc0 + 32, hi), s0);
      const _Float16* kc1 = Kbuf + (size_t)(16 + col) * D_H;
      s1 = wmma16x16x32(aq0, load_bfrag_f16(kc1, hi), s1);
      s1 = wmma16x16x32(aq1, load_bfrag_f16(kc1 + 32, hi), s1);
    }
    const int mv0 = mrow[key0 + col];
    const int mv1 = mrow[key0 + 16 + col];

    // ---- online softmax over this 32-key block ---------------------------
    float p0[8], p1[8];
#pragma unroll
    for (int r = 0; r < 8; ++r) {
      float x0 = mv0 ? s0[r] * scale : -1e30f;
      float x1 = mv1 ? s1[r] * scale : -1e30f;
      float bm = fmaxf(x0, x1);
      bm = fmaxf(bm, __shfl_xor(bm, 1, 32));
      bm = fmaxf(bm, __shfl_xor(bm, 2, 32));
      bm = fmaxf(bm, __shfl_xor(bm, 4, 32));
      bm = fmaxf(bm, __shfl_xor(bm, 8, 32));   // row max within 16-lane half
      float mnew  = fmaxf(mi[r], bm);
      float alpha = __expf(mi[r] - mnew);
      float e0 = __expf(x0 - mnew);
      float e1 = __expf(x1 - mnew);
      float rs = e0 + e1;
      rs += __shfl_xor(rs, 1, 32);
      rs += __shfl_xor(rs, 2, 32);
      rs += __shfl_xor(rs, 4, 32);
      rs += __shfl_xor(rs, 8, 32);             // row sum
      li[r] = li[r] * alpha + rs;
      mi[r] = mnew;
      p0[r] = e0; p1[r] = e1;
#pragma unroll
      for (int ct = 0; ct < 4; ++ct) acc[ct][r] *= alpha;
    }

    // ---- stage P (16x32) in LDS, reload in A-frag layout -----------------
#pragma unroll
    for (int r = 0; r < 8; ++r) {
      pb[(hi * 8 + r) * 32 + col]      = (_Float16)p0[r];
      pb[(hi * 8 + r) * 32 + 16 + col] = (_Float16)p1[r];
    }
    asm volatile("s_wait_dscnt 0" ::: "memory");
    v16h aP = load_afrag_f16(pb + col * 32, 0, hi);

    // ---- O += P @ V (V from LDS) -----------------------------------------
#pragma unroll
    for (int ct = 0; ct < 4; ++ct)
      acc[ct] = wmma16x16x32(aP, load_bfrag_f16(Vbuf + (ct * 16 + col) * 32, hi),
                             acc[ct]);
  }

  // ---- normalize and store to attn16 [N*L][E] ----------------------------
#pragma unroll
  for (int ct = 0; ct < 4; ++ct) {
#pragma unroll
    for (int r = 0; r < 8; ++r) {
      float o = acc[ct][r] / li[r];
      attn16[(size_t)(n * L_S + l0 + hi * 8 + r) * E_D + h * D_H + ct * 16 + col] =
          (_Float16)o;
    }
  }
}

// ===========================================================================
// Kernel 4: out = attn16 @ Wo16^T + bo  (rows = N*L = 4096, fp32 out)
// Block = 4 waves on 4 row-tiles of the SAME 64-col group; the Wo16 column
// block (64 x 32 f16 per K-step) is async-staged in LDS, shared by 4 waves.
// ===========================================================================
__global__ void __launch_bounds__(128)
outproj_kernel(const _Float16* __restrict__ a16, const _Float16* __restrict__ Wo16,
               const float* __restrict__ bo, float* __restrict__ out) {
  __shared__ _Float16 Wbuf[64 * 32];      // [j-local][k-local]

  const int tid   = threadIdx.x;
  const int wslot = tid >> 5;
  const int lane  = tid & 31;
  const int col   = lane & 15, hi = lane >> 4;

  const int b  = blockIdx.x;              // 1024 blocks: rg(64) x cg(16)
  const int cg = b & 15;
  const int rg = b >> 4;
  const int rt = rg * 4 + wslot;          // this wave's 16-row tile (0..255)
  const int j0 = cg * 64;

  v8f acc[4];
#pragma unroll
  for (int ct = 0; ct < 4; ++ct) {
    float bv = bo[j0 + ct * 16 + col];    // bias depends only on column (= lane)
#pragma unroll
    for (int r = 0; r < 8; ++r) acc[ct][r] = bv;
  }

  const _Float16* arow = a16 + (size_t)(rt * 16 + col) * E_D;
  const int wj = tid >> 1, wp = tid & 1;
  const unsigned wl_off = lds_offset(Wbuf) + (wj * 32 + wp * 16) * 2;   // bytes
  const _Float16* wsrc = Wo16 + (size_t)(j0 + wj) * E_D + wp * 16;

  for (int ks = 0; ks < E_D; ks += 32) {
    __syncthreads();
    async_copy_b128(wl_off,      wsrc + ks);
    async_copy_b128(wl_off + 16, wsrc + ks + 8);
    wait_async0();
    __syncthreads();

    v16h a = load_afrag_f16(arow, ks, hi);
#pragma unroll
    for (int ct = 0; ct < 4; ++ct)
      acc[ct] = wmma16x16x32(a, load_bfrag_f16(Wbuf + (ct * 16 + col) * 32, hi),
                             acc[ct]);
  }

#pragma unroll
  for (int ct = 0; ct < 4; ++ct)
#pragma unroll
    for (int r = 0; r < 8; ++r)
      out[(size_t)(rt * 16 + hi * 8 + r) * E_D + j0 + ct * 16 + col] = acc[ct][r];
}

// ===========================================================================
extern "C" void kernel_launch(void* const* d_in, const int* in_sizes, int n_in,
                              void* d_out, int out_size, void* d_ws, size_t ws_size,
                              hipStream_t stream) {
  (void)in_sizes; (void)n_in; (void)out_size; (void)ws_size;
  const float* keys    = (const float*)d_in[0];
  const float* queries = (const float*)d_in[1];
  /* d_in[2] = values: unused (reference computes v = q @ Wv^T) */
  const int*   mask    = (const int*)d_in[3];
  const float* Wk      = (const float*)d_in[4];
  const float* Wq      = (const float*)d_in[5];
  const float* Wv      = (const float*)d_in[6];
  const float* Wo      = (const float*)d_in[7];
  const float* bo      = (const float*)d_in[8];
  float* out = (float*)d_out;

  const size_t HE = (size_t)N_B * H_N * L_S * D_H;   // 4,194,304 elements
  _Float16* q16  = (_Float16*)d_ws;
  _Float16* k16  = q16 + HE;
  _Float16* vT   = k16 + HE;
  _Float16* a16  = vT + HE;                           // [N*L][E] f16
  _Float16* Wo16 = a16 + HE;                          // 1M f16
  _Float16* Wq16 = Wo16 + (size_t)E_D * E_D;
  _Float16* Wk16 = Wq16 + D_H * D_H;
  _Float16* Wv16 = Wk16 + D_H * D_H;

  dim3 blk(128);
  // ---- one-time weight conversions (f32 -> f16) --------------------------
  cvt_f16_kernel<<<(E_D * E_D) / 1024, 256, 0, stream>>>(Wo, Wo16, E_D * E_D);
  cvt_f16_kernel<<<4, 256, 0, stream>>>(Wq, Wq16, D_H * D_H);
  cvt_f16_kernel<<<4, 256, 0, stream>>>(Wk, Wk16, D_H * D_H);
  cvt_f16_kernel<<<4, 256, 0, stream>>>(Wv, Wv16, D_H * D_H);

  // ---- pipeline (4096 wave-tasks per stage -> 1024 blocks, exact cover) --
  proj_f32_kernel<<<1024, blk, 0, stream>>>(queries, Wq16, q16);
  proj_f32_kernel<<<1024, blk, 0, stream>>>(keys,    Wk16, k16);
  vproj_kernel   <<<1024, blk, 0, stream>>>(q16, Wv16, vT);
  attn_kernel    <<<1024, blk, 0, stream>>>(q16, k16, vT, mask, a16);
  outproj_kernel <<<1024, blk, 0, stream>>>(a16, Wo16, bo, out);
}
